// RNN_46643344835012
// MI455X (gfx1250) — compile-verified, matrix operand order checked by reading, not run on hardware
//
#include <hip/hip_runtime.h>

// ---- problem dims ----
#define Bt   64
#define Tt   200
#define NSK  1024   // num_skills
#define CN   256    // concept_num
#define HID  1024   // hidden
#define IN2  512    // 2*concept_num (LSTM input)
#define G4   4096   // 4*hidden (gates)
#define MROWS (Bt * Tt)   // 12800

typedef __bf16 bf16;
typedef __attribute__((ext_vector_type(16))) __bf16 v16bf;
typedef __attribute__((ext_vector_type(8)))  __bf16 v8bf;
typedef __attribute__((ext_vector_type(8)))  float  v8f;
typedef __attribute__((ext_vector_type(4)))  float  v4f;

// ---- WMMA (gfx1250 wave32, 16x16x32 bf16 -> f32 accum) ----
__device__ __forceinline__ v8f wmma_bf16(v16bf a, v16bf b, v8f c) {
  return __builtin_amdgcn_wmma_f32_16x16x32_bf16(false, a, false, b, (short)0, c,
                                                 false, false);
}

// CDNA5 async global->LDS copy (VGLOBAL encoding, tracked by ASYNCcnt).
// vdst = per-lane LDS byte offset, vaddr = 64-bit global address.
__device__ __forceinline__ void async_cp_b128(unsigned lds_off, const bf16* gaddr) {
  asm volatile("global_load_async_to_lds_b128 %0, %1, off"
               :: "v"(lds_off), "v"(gaddr)
               : "memory");
}
__device__ __forceinline__ void async_wait0() {
  asm volatile("s_wait_asynccnt 0x0" ::: "memory");
}

// A/B fragment loader from row-major [rows, K] bf16 (global or LDS).
// 16-bit A layout: lane = (kgroup<<4) | m ; VGPR0..3 -> K = kg*8+0..7,
// VGPR4..7 -> K = 16+kg*8+0..7  => two contiguous 16B chunks per lane.
__device__ __forceinline__ v16bf load_frag_bf16(const bf16* __restrict__ base,
                                                int ld, int row0, int k0) {
  const int lane = threadIdx.x & 31;
  const int r = lane & 15, kg = lane >> 4;
  const bf16* p = base + (size_t)(row0 + r) * ld + k0 + kg * 8;
  v8bf lo = *reinterpret_cast<const v8bf*>(p);
  v8bf hi = *reinterpret_cast<const v8bf*>(p + 16);
  v16bf o;
#pragma unroll
  for (int i = 0; i < 8; ++i) { o[i] = lo[i]; o[i + 8] = hi[i]; }
  return o;
}

// Same fragment but fp32 source, converted to bf16 in-register.
__device__ __forceinline__ v16bf load_frag_f32(const float* __restrict__ base,
                                               int ld, int row0, int k0) {
  const int lane = threadIdx.x & 31;
  const int r = lane & 15, kg = lane >> 4;
  const float* p = base + (size_t)(row0 + r) * ld + k0 + kg * 8;
  v4f a0 = *reinterpret_cast<const v4f*>(p);
  v4f a1 = *reinterpret_cast<const v4f*>(p + 4);
  v4f b0 = *reinterpret_cast<const v4f*>(p + 16);
  v4f b1 = *reinterpret_cast<const v4f*>(p + 20);
  v16bf o;
#pragma unroll
  for (int i = 0; i < 4; ++i) {
    o[i]      = (bf16)a0[i];
    o[i + 4]  = (bf16)a1[i];
    o[i + 8]  = (bf16)b0[i];
    o[i + 12] = (bf16)b1[i];
  }
  return o;
}

// ---- K1: fp32 -> bf16 weight conversion ----
__global__ void cvt_f32_bf16(const float* __restrict__ src, bf16* __restrict__ dst, int n) {
  int i = blockIdx.x * blockDim.x + threadIdx.x;
  int stride = gridDim.x * blockDim.x;
  for (; i < n; i += stride) dst[i] = (bf16)src[i];
}

// ---- K2: kc GEMM. A = q fp32 [12800,1024], B = W_kc bf16 [256,1024].
// mode 0: scatter into kc_vec bf16 [12800,512] per result mask (fused concat)
// mode 1: next_kc fp32 [12800,256]
__global__ void kc_gemm(const float* __restrict__ A, const bf16* __restrict__ Wb,
                        const float* __restrict__ bias, const int* __restrict__ result,
                        bf16* __restrict__ kc_vec, float* __restrict__ out_f32, int mode) {
  const int wave = threadIdx.x >> 5;
  const int lane = threadIdx.x & 31;
  const int mtile = blockIdx.x * 8 + wave;   // 0..799
  const int n0 = blockIdx.y * 64;            // 4 n-tiles per wave
  v8f acc[4] = {};
  for (int kk = 0; kk < NSK; kk += 32) {
    v16bf a = load_frag_f32(A, NSK, mtile * 16, kk);
#pragma unroll
    for (int nt = 0; nt < 4; ++nt) {
      v16bf b = load_frag_bf16(Wb, NSK, n0 + nt * 16, kk);
      acc[nt] = wmma_bf16(a, b, acc[nt]);
    }
  }
  const int nl = lane & 15, rh = lane >> 4;
#pragma unroll
  for (int nt = 0; nt < 4; ++nt) {
    int n = n0 + nt * 16 + nl;
    float bv = bias[n];
#pragma unroll
    for (int v = 0; v < 8; ++v) {
      int row = mtile * 16 + v + 8 * rh;     // C layout: M = v + 8*(lane>>4)
      float val = acc[nt][v] + bv;
      if (mode == 0) {
        int rp = result[row];
        bf16* dst = kc_vec + (size_t)row * IN2;
        if (rp == 1) { dst[n] = (bf16)val;  dst[CN + n] = (bf16)0.0f; }
        else         { dst[n] = (bf16)0.0f; dst[CN + n] = (bf16)val;  }
      } else {
        out_f32[(size_t)row * CN + n] = val;
      }
    }
  }
}

// ---- K3: gx = kc_vec @ W_ih^T + (b_ih + b_hh), bf16 out [12800, 4096] ----
__global__ void gx_gemm(const bf16* __restrict__ kc_vec, const bf16* __restrict__ Wih,
                        const float* __restrict__ b_ih, const float* __restrict__ b_hh,
                        bf16* __restrict__ gx) {
  const int wave = threadIdx.x >> 5;
  const int lane = threadIdx.x & 31;
  const int mtile = blockIdx.x * 8 + wave;
  const int n0 = blockIdx.y * 64;
  v8f acc[4] = {};
  for (int kk = 0; kk < IN2; kk += 32) {
    v16bf a = load_frag_bf16(kc_vec, IN2, mtile * 16, kk);
#pragma unroll
    for (int nt = 0; nt < 4; ++nt) {
      v16bf b = load_frag_bf16(Wih, IN2, n0 + nt * 16, kk);
      acc[nt] = wmma_bf16(a, b, acc[nt]);
    }
  }
  const int nl = lane & 15, rh = lane >> 4;
#pragma unroll
  for (int nt = 0; nt < 4; ++nt) {
    int n = n0 + nt * 16 + nl;
    float bv = b_ih[n] + b_hh[n];
#pragma unroll
    for (int v = 0; v < 8; ++v) {
      int row = mtile * 16 + v + 8 * rh;
      gx[(size_t)row * G4 + n] = (bf16)(acc[nt][v] + bv);
    }
  }
}

// ---- K4: persistent LSTM. One launch, 200 steps, device-wide step barrier.
// 32 blocks; block j owns hidden [j*32, j*32+32) across ALL FOUR gates.
// W_hh slice (256 KB) staged to LDS ONCE and served via ds_load for all steps;
// h_prev staged per 128-wide K-chunk via async global->LDS, double-buffered.
// Gate-exchange buffer overlays the h staging buffers (disjoint lifetimes).
// Cell state c lives in registers. Total LDS = 256K + 32K = 288 KB.
__global__ void lstm_persistent(const bf16* __restrict__ gx, const bf16* __restrict__ Whh,
                                bf16* __restrict__ h_buf, bf16* __restrict__ h_seq,
                                unsigned* __restrict__ bar) {
  __shared__ bf16 lds_whh[128 * HID];    // 256 KB: block's 128 gate-columns x K
  __shared__ char lds_un[32 * 1024];     // 32 KB: h staging (2x16K) U gate xchg
  bf16*  lds_h = (bf16*)lds_un;          // [2][Bt*128]
  float* lds_g = (float*)lds_un;         // [4][Bt][32]

  const int tid  = threadIdx.x;
  const int wave = tid >> 5, lane = tid & 31;
  const int j    = blockIdx.x;           // hidden chunk
  const int gate = wave >> 1, sub = wave & 1;
  const int ng   = gate * HID + j * 32 + sub * 16;   // global gate column base
  const int nl   = lane & 15, rh = lane >> 4;

  // ---- one-time: stage this block's W_hh slice into LDS ----
  // local row lrow = gate*32 + c  (c = column within hidden chunk), K-major.
  {
    int lrow = tid >> 1;                 // 0..127
    int cb   = (tid & 1) * 512;          // half-row of K
    int g    = lrow >> 5, c = lrow & 31;
    const bf16* gsrc = Whh + ((size_t)(g * HID + j * 32 + c)) * HID + cb;
    unsigned loff = (unsigned)(size_t)(const void*)&lds_whh[lrow * HID + cb];
#pragma unroll 8
    for (int i = 0; i < 64; ++i) async_cp_b128(loff + i * 16, gsrc + i * 8);
  }
  async_wait0();
  __syncthreads();

  // staging geometry for h chunks: thread copies one 32-elem row segment
  const int srow = tid >> 2;             // 0..63  (batch row)
  const int scol = (tid & 3) * 32;       // 0,32,64,96 within 128-wide chunk

  // persistent cell state: element e = tid + k*256 -> (b = e>>5, hl = e&31)
  float creg[8];
#pragma unroll
  for (int k = 0; k < 8; ++k) creg[k] = 0.f;

  for (int t = 0; t < Tt; ++t) {
    const bf16* h_prev = h_buf + (size_t)(t & 1) * (Bt * HID);
    bf16*       h_next = h_buf + (size_t)((t + 1) & 1) * (Bt * HID);

    // prefetch next step's gx tile for this block
    if (t + 1 < Tt) {
      const bf16* pf = gx + ((size_t)srow * Tt + (t + 1)) * G4 + (tid & 3) * HID + j * 32;
      __builtin_prefetch(pf, 0, 3);
    }

    // stage chunk 0 (cols 0..127) of h_prev
    {
      unsigned loff = (unsigned)(size_t)(const void*)&lds_h[srow * 128 + scol];
      const bf16* gsrc = h_prev + (size_t)srow * HID + scol;
#pragma unroll
      for (int i = 0; i < 4; ++i) async_cp_b128(loff + i * 16, gsrc + i * 8);
    }
    async_wait0();
    __syncthreads();

    v8f acc[4] = {};
    for (int kc = 0; kc < 8; ++kc) {
      if (kc + 1 < 8) {   // overlap: fill next buffer while computing
        unsigned loff = (unsigned)(size_t)(const void*)
            &lds_h[(((kc + 1) & 1) * Bt + srow) * 128 + scol];
        const bf16* gsrc = h_prev + (size_t)srow * HID + (kc + 1) * 128 + scol;
#pragma unroll
        for (int i = 0; i < 4; ++i) async_cp_b128(loff + i * 16, gsrc + i * 8);
      }
      const bf16* lh = lds_h + (size_t)(kc & 1) * (Bt * 128);
#pragma unroll
      for (int ks = 0; ks < 4; ++ks) {
        // B fragment from resident LDS copy of W_hh (row0 = wave*16 local cols)
        v16bf bfrag = load_frag_bf16(lds_whh, HID, wave * 16, kc * 128 + ks * 32);
#pragma unroll
        for (int mt = 0; mt < 4; ++mt) {
          v16bf a = load_frag_bf16(lh, 128, mt * 16, ks * 32);
          acc[mt] = wmma_bf16(a, bfrag, acc[mt]);
        }
      }
      async_wait0();
      __syncthreads();
    }

    // add gx preactivations, exchange gates via LDS (overlays h staging)
#pragma unroll
    for (int mt = 0; mt < 4; ++mt) {
#pragma unroll
      for (int v = 0; v < 8; ++v) {
        int b = mt * 16 + v + 8 * rh;
        float pre = acc[mt][v] + (float)gx[((size_t)b * Tt + t) * G4 + ng + nl];
        lds_g[((gate * Bt) + b) * 32 + sub * 16 + nl] = pre;
      }
    }
    __syncthreads();

#pragma unroll
    for (int k = 0; k < 8; ++k) {
      int e = tid + k * 256;
      int b = e >> 5, hl = e & 31;
      int hidx = j * 32 + hl;
      float gi = lds_g[(0 * Bt + b) * 32 + hl];
      float gf = lds_g[(1 * Bt + b) * 32 + hl];
      float gg = lds_g[(2 * Bt + b) * 32 + hl];
      float go = lds_g[(3 * Bt + b) * 32 + hl];
      float si = 1.f / (1.f + __expf(-gi));
      float sf = 1.f / (1.f + __expf(-gf));
      float tg = tanhf(gg);
      float so = 1.f / (1.f + __expf(-go));
      float cc = sf * creg[k] + si * tg;
      creg[k] = cc;
      bf16 hb = (bf16)(so * tanhf(cc));
      h_next[(size_t)b * HID + hidx] = hb;
      h_seq[((size_t)b * Tt + t) * HID + hidx] = hb;
    }
    __syncthreads();   // gate reads done before next step reuses union LDS

    // device-wide step barrier (32 resident blocks)
    if (tid == 0) {
      __threadfence();
      __hip_atomic_fetch_add(bar, 1u, __ATOMIC_RELEASE, __HIP_MEMORY_SCOPE_AGENT);
      unsigned target = 32u * (unsigned)(t + 1);
      while (__hip_atomic_load(bar, __ATOMIC_ACQUIRE, __HIP_MEMORY_SCOPE_AGENT) < target)
        __builtin_amdgcn_s_sleep(1);
    }
    __syncthreads();
  }
}

// ---- K5: stu_state = h_seq @ W_state^T + b_state, fp32 out ----
__global__ void stu_gemm(const bf16* __restrict__ h_seq, const bf16* __restrict__ Wst,
                         const float* __restrict__ b_state, float* __restrict__ stu) {
  const int wave = threadIdx.x >> 5;
  const int lane = threadIdx.x & 31;
  const int mtile = blockIdx.x * 8 + wave;
  const int n0 = blockIdx.y * 64;
  v8f acc[4] = {};
  for (int kk = 0; kk < HID; kk += 32) {
    v16bf a = load_frag_bf16(h_seq, HID, mtile * 16, kk);
#pragma unroll
    for (int nt = 0; nt < 4; ++nt) {
      v16bf b = load_frag_bf16(Wst, HID, n0 + nt * 16, kk);
      acc[nt] = wmma_bf16(a, b, acc[nt]);
    }
  }
  const int nl = lane & 15, rh = lane >> 4;
#pragma unroll
  for (int nt = 0; nt < 4; ++nt) {
    int n = n0 + nt * 16 + nl;
    float bv = b_state[n];
#pragma unroll
    for (int v = 0; v < 8; ++v) {
      int row = mtile * 16 + v + 8 * rh;
      stu[(size_t)row * CN + n] = acc[nt][v] + bv;
    }
  }
}

// ---- K6: res = ((stu - nk) * nk) @ W_out^T + b_out, one wave32 per row ----
__global__ void res_kernel(const float* __restrict__ stu, const float* __restrict__ nk,
                           const float* __restrict__ W_out, const float* __restrict__ b_out,
                           float* __restrict__ res) {
  const int wave = threadIdx.x >> 5;
  const int lane = threadIdx.x & 31;
  const int row = blockIdx.x * 8 + wave;
  float p = 0.f;
#pragma unroll
  for (int k = 0; k < 8; ++k) {
    int n = lane + k * 32;
    float s = stu[(size_t)row * CN + n];
    float q = nk[(size_t)row * CN + n];
    p += (s - q) * q * W_out[n];
  }
#pragma unroll
  for (int off = 16; off > 0; off >>= 1) p += __shfl_down(p, off, 32);
  if (lane == 0) res[row] = p + b_out[0];
}

extern "C" void kernel_launch(void* const* d_in, const int* in_sizes, int n_in,
                              void* d_out, int out_size, void* d_ws, size_t ws_size,
                              hipStream_t stream) {
  (void)in_sizes; (void)n_in; (void)out_size; (void)ws_size;
  const float* q_hot   = (const float*)d_in[0];
  const int*   result  = (const int*)d_in[1];
  const float* next_q  = (const float*)d_in[2];
  // d_in[3] = concept_num scalar (compile-time constant here)
  const float* W_kc    = (const float*)d_in[4];
  const float* b_kc    = (const float*)d_in[5];
  const float* W_ih    = (const float*)d_in[6];
  const float* W_hh    = (const float*)d_in[7];
  const float* b_ih    = (const float*)d_in[8];
  const float* b_hh    = (const float*)d_in[9];
  const float* W_state = (const float*)d_in[10];
  const float* b_state = (const float*)d_in[11];
  const float* W_out   = (const float*)d_in[12];
  const float* b_out   = (const float*)d_in[13];

  char* ws = (char*)d_ws;
  size_t off = 0;
  auto take = [&](size_t bytes) {
    char* p = ws + off;
    off = (off + bytes + 255) & ~(size_t)255;
    return p;
  };
  bf16*     Wkc_b  = (bf16*)take((size_t)CN * NSK * 2);
  bf16*     Wih_b  = (bf16*)take((size_t)G4 * IN2 * 2);
  bf16*     Whh_b  = (bf16*)take((size_t)G4 * HID * 2);
  bf16*     Wst_b  = (bf16*)take((size_t)CN * HID * 2);
  bf16*     kc_vec = (bf16*)take((size_t)MROWS * IN2 * 2);
  float*    nkc    = (float*)take((size_t)MROWS * CN * 4);
  bf16*     gx     = (bf16*)take((size_t)MROWS * G4 * 2);
  bf16*     h_seq  = (bf16*)take((size_t)MROWS * HID * 2);
  bf16*     h_buf  = (bf16*)take((size_t)2 * Bt * HID * 2);
  unsigned* bar    = (unsigned*)take(256);

  float* res_out = (float*)d_out;            // [12800] first
  float* stu_out = (float*)d_out + MROWS;    // then [12800, 256]

  // zero h0 and the step-barrier counter (captured ops -> reset every replay)
  hipMemsetAsync(h_buf, 0, (size_t)Bt * HID * 2, stream);
  hipMemsetAsync(bar, 0, 256, stream);

  dim3 blk(256);
  cvt_f32_bf16<<<512,  blk, 0, stream>>>(W_kc,    Wkc_b, CN * NSK);
  cvt_f32_bf16<<<1024, blk, 0, stream>>>(W_ih,    Wih_b, G4 * IN2);
  cvt_f32_bf16<<<2048, blk, 0, stream>>>(W_hh,    Whh_b, G4 * HID);
  cvt_f32_bf16<<<512,  blk, 0, stream>>>(W_state, Wst_b, CN * HID);

  kc_gemm<<<dim3(100, 4), blk, 0, stream>>>(q_hot, Wkc_b, b_kc, result,
                                            kc_vec, nullptr, 0);
  kc_gemm<<<dim3(100, 4), blk, 0, stream>>>(next_q, Wkc_b, b_kc, nullptr,
                                            nullptr, nkc, 1);
  gx_gemm<<<dim3(100, 64), blk, 0, stream>>>(kc_vec, Wih_b, b_ih, b_hh, gx);

  lstm_persistent<<<32, blk, 0, stream>>>(gx, Whh_b, h_buf, h_seq, bar);

  stu_gemm<<<dim3(100, 4), blk, 0, stream>>>(h_seq, Wst_b, b_state, stu_out);
  res_kernel<<<1600, blk, 0, stream>>>(stu_out, nkc, W_out, b_out, res_out);
}
